// CausalFlowAttention_84593675862634
// MI455X (gfx1250) — matrix operand order repaired and checked by reading, not compiled, not run on hardware
//
#include <hip/hip_runtime.h>

// ---------------------------------------------------------------------------
// CausalFlowAttention for MI455X (gfx1250, wave32, WMMA 16x16x32 bf16)
// Mask = causal | same_chunk(64)  ==>  block-causal with FULL 64x64 blocks:
// query chunk cq attends to key chunks 0..cq with no element masking.
//
// Pipeline:
//   0) one-shot fp32 -> bf16 conversion of x, Wqkv, Wout (keeps GEMM inner
//      loops free of v_cvt and halves L2 read traffic)
//   1) qkv_gemm   : bf16 WMMA GEMM, wave tile 32x128, scatter epilogue to
//                   Q[b,h,t,d], K[b,h,t,d], Vt[b,h,d,t]
//   2) attn_kernel: flash attention over full 64-key blocks, WMMA for QK^T
//                   and PV, LDS round-trip for C->A layout of P
//   3) out_gemm   : bf16 WMMA GEMM -> fp32 output
// ---------------------------------------------------------------------------

typedef __attribute__((ext_vector_type(16))) __bf16 v16bf;
typedef __attribute__((ext_vector_type(8)))  float  v8f;
typedef __attribute__((ext_vector_type(8)))  __bf16 bf16x8;
typedef __attribute__((ext_vector_type(4)))  float  f32x4;

#define DEVINL static __device__ __forceinline__

#define BATCH 2
#define SEQ   2048
#define DIM   1024
#define HEADS 16
#define HD    64
#define CHUNK 64

DEVINL v8f wmma_bf16(v16bf a, v16bf b, v8f c) {
  // v_wmma_f32_16x16x32_bf16  D = A(16x32) * B(32x16) + C
  return __builtin_amdgcn_wmma_f32_16x16x32_bf16(false, a, false, b, (short)0, c,
                                                 false, false);
}

// A fragment (16x32) from bf16 row-major [., lda] (global or LDS).
// A-layout: lane L holds row (L&15); halves hold K {hs*8..+7, 16+hs*8..+7}.
DEVINL v16bf a_frag_bf16(const __bf16* __restrict__ A, int lda, int row0, int k0) {
  const int lane = threadIdx.x & 31;
  const int hs = lane >> 4;
  const __bf16* p = A + (size_t)(row0 + (lane & 15)) * lda + k0 + hs * 8;
  bf16x8 lo = *(const bf16x8*)(p);
  bf16x8 hi = *(const bf16x8*)(p + 16);
  v16bf a;
#pragma unroll
  for (int e = 0; e < 8; ++e) { a[e] = lo[e]; a[8 + e] = hi[e]; }
  return a;
}

// B fragment (32x16) with B[k,n] = S[n,k]; S bf16 row-major [N, ldk].
// B-layout: lane L holds col (L&15); halves hold K {hs*8..+7, 16+hs*8..+7}.
DEVINL v16bf b_frag_bf16T(const __bf16* __restrict__ S, int ldk, int n0, int k0) {
  const int lane = threadIdx.x & 31;
  const int hs = lane >> 4;
  const __bf16* p = S + (size_t)(n0 + (lane & 15)) * ldk + k0 + hs * 8;
  bf16x8 lo = *(const bf16x8*)(p);
  bf16x8 hi = *(const bf16x8*)(p + 16);
  v16bf b;
#pragma unroll
  for (int e = 0; e < 8; ++e) { b[e] = lo[e]; b[8 + e] = hi[e]; }
  return b;
}

// ---------------------------------------------------------------------------
// Kernel 0: one-shot fp32 -> bf16 (n must be a multiple of 8*256 per launch)
// ---------------------------------------------------------------------------
__global__ __launch_bounds__(256) void cvt_f32_to_bf16(
    const float* __restrict__ src, __bf16* __restrict__ dst, int n) {
  const int i = (blockIdx.x * 256 + threadIdx.x) * 8;
  if (i < n) {
    f32x4 a = *(const f32x4*)(src + i);
    f32x4 b = *(const f32x4*)(src + i + 4);
    bf16x8 o;
#pragma unroll
    for (int e = 0; e < 4; ++e) { o[e] = (__bf16)a[e]; o[4 + e] = (__bf16)b[e]; }
    *(bf16x8*)(dst + i) = o;
  }
}

// ---------------------------------------------------------------------------
// Kernel 1: QKV projection. C[M=4096, N=3072] = xbf * Wqkvbf^T.
// Wave tile 32x128 (2x8 WMMA tiles); WG = 4 waves covering 64x256.
// Epilogue scatters q->Q[b,h,t,d], k->K[b,h,t,d], v->Vt[b,h,d,t].
// ---------------------------------------------------------------------------
__global__ __launch_bounds__(128) void qkv_gemm(
    const __bf16* __restrict__ x, const __bf16* __restrict__ Wqkv,
    __bf16* __restrict__ Qb, __bf16* __restrict__ Kb, __bf16* __restrict__ Vt) {
  const int w    = threadIdx.x >> 5;
  const int lane = threadIdx.x & 31;
  const int r0 = blockIdx.x * 64 + (w & 1) * 32;
  const int c0 = blockIdx.y * 256 + (w >> 1) * 128;

  const v8f zero = {0.f, 0.f, 0.f, 0.f, 0.f, 0.f, 0.f, 0.f};
  v8f acc[2][8];
#pragma unroll
  for (int rt = 0; rt < 2; ++rt)
#pragma unroll
    for (int ct = 0; ct < 8; ++ct) acc[rt][ct] = zero;

  for (int k0 = 0; k0 < DIM; k0 += 32) {
    v16bf a0 = a_frag_bf16(x, DIM, r0, k0);
    v16bf a1 = a_frag_bf16(x, DIM, r0 + 16, k0);
#pragma unroll
    for (int ct = 0; ct < 8; ++ct) {
      v16bf b = b_frag_bf16T(Wqkv, DIM, c0 + ct * 16, k0);
      acc[0][ct] = wmma_bf16(a0, b, acc[0][ct]);
      acc[1][ct] = wmma_bf16(a1, b, acc[1][ct]);
    }
  }

  const int hs = lane >> 4;
#pragma unroll
  for (int rt = 0; rt < 2; ++rt) {
#pragma unroll
    for (int ct = 0; ct < 8; ++ct) {
      const int n0  = c0 + ct * 16;       // global col, multiple of 16
      const int s   = n0 >> 10;           // 0=q 1=k 2=v
      const int rem = n0 & (DIM - 1);
      const int h   = rem >> 6;
      const int d   = (rem & 63) + (lane & 15);
#pragma unroll
      for (int i = 0; i < 8; ++i) {
        const int t  = r0 + rt * 16 + hs * 8 + i;   // global row (B*T)
        const int b  = t >> 11;
        const int tl = t & (SEQ - 1);
        const size_t bh = (size_t)(b * HEADS + h);
        const __bf16 val = (__bf16)acc[rt][ct][i];
        if (s == 0)      Qb[(bh * SEQ + tl) * HD + d] = val;
        else if (s == 1) Kb[(bh * SEQ + tl) * HD + d] = val;
        else             Vt[(bh * HD + d) * SEQ + tl] = val;
      }
    }
  }
}

// ---------------------------------------------------------------------------
// Kernel 2: flash attention, one WG per (b, h, query-chunk of 64).
// 4 waves x 16 query rows. Key blocks of 64, kb = 0..cq, fully visible.
// ---------------------------------------------------------------------------
__global__ __launch_bounds__(128) void attn_kernel(
    const __bf16* __restrict__ Qb, const __bf16* __restrict__ Kb,
    const __bf16* __restrict__ Vt, __bf16* __restrict__ AO) {
  __shared__ __align__(16) __bf16 lds[4 * 16 * 80];   // per-wave 16x64 P, stride 80

  const int w    = threadIdx.x >> 5;
  const int lane = threadIdx.x & 31;
  const int hs   = lane >> 4;
  const int cq = blockIdx.x;      // 0..31
  const int h  = blockIdx.y;      // 0..15
  const int b  = blockIdx.z;      // 0..1
  const size_t bh = (size_t)(b * HEADS + h);

  const __bf16* Qh = Qb + bh * SEQ * HD;   // [T, hd]
  const __bf16* Kh = Kb + bh * SEQ * HD;   // [T, hd]
  const __bf16* Vh = Vt + bh * HD * SEQ;   // [hd, T]
  __bf16* P = &lds[w * 16 * 80];

  const int q0 = cq * CHUNK + w * 16;      // this wave's 16 query rows
  const float scale = 0.125f;              // 1/sqrt(64)

  v16bf qf0 = a_frag_bf16(Qh, HD, q0, 0);
  v16bf qf1 = a_frag_bf16(Qh, HD, q0, 32);

  const v8f zero = {0.f, 0.f, 0.f, 0.f, 0.f, 0.f, 0.f, 0.f};
  v8f o[4];
  float m[8], l[8];
#pragma unroll
  for (int dt = 0; dt < 4; ++dt) o[dt] = zero;
#pragma unroll
  for (int i = 0; i < 8; ++i) { m[i] = -1e30f; l[i] = 0.f; }

  for (int kb = 0; kb <= cq; ++kb) {
    const int j0 = kb * CHUNK;

    // ---- prefetch next key/value block (uniform branch; EXEC stays full) --
    if (kb < cq) {
      const int jn = j0 + CHUNK;
      __builtin_prefetch(Kh + (size_t)jn * HD + lane * 128, 0, 3);       // 8KB K blk
      __builtin_prefetch(Vh + (size_t)lane * SEQ + jn, 0, 3);            // V rows
      __builtin_prefetch(Vh + (size_t)(lane + 32) * SEQ + jn, 0, 3);
    }

    // ---- S = scale * Q K^T  (4 column tiles of 16 keys) ----
    v8f s4[4];
#pragma unroll
    for (int nt = 0; nt < 4; ++nt) {
      v16bf bk0 = b_frag_bf16T(Kh, HD, j0 + nt * 16, 0);
      v16bf bk1 = b_frag_bf16T(Kh, HD, j0 + nt * 16, 32);
      v8f s = zero;
      s = wmma_bf16(qf0, bk0, s);
      s = wmma_bf16(qf1, bk1, s);
#pragma unroll
      for (int i = 0; i < 8; ++i) s[i] *= scale;
      s4[nt] = s;
    }

    // ---- row max across 64 keys (lanes 0-15 hold rows 0-7, 16-31 rows 8-15)
    float rm[8];
#pragma unroll
    for (int i = 0; i < 8; ++i) {
      float v = fmaxf(fmaxf(s4[0][i], s4[1][i]), fmaxf(s4[2][i], s4[3][i]));
#pragma unroll
      for (int d = 1; d < 16; d <<= 1) v = fmaxf(v, __shfl_xor(v, d, 32));
      rm[i] = v;
    }

    // ---- online softmax update ----
    float al[8], rs[8];
#pragma unroll
    for (int i = 0; i < 8; ++i) {
      const float mn = fmaxf(m[i], rm[i]);
      al[i] = __expf(m[i] - mn);
      m[i] = mn;
      rs[i] = 0.f;
    }
#pragma unroll
    for (int nt = 0; nt < 4; ++nt)
#pragma unroll
      for (int i = 0; i < 8; ++i) {
        const float p = __expf(s4[nt][i] - m[i]);
        s4[nt][i] = p;
        rs[i] += p;
      }
#pragma unroll
    for (int i = 0; i < 8; ++i) {
#pragma unroll
      for (int d = 1; d < 16; d <<= 1) rs[i] += __shfl_xor(rs[i], d, 32);
      l[i] = l[i] * al[i] + rs[i];
    }
#pragma unroll
    for (int dt = 0; dt < 4; ++dt)
#pragma unroll
      for (int i = 0; i < 8; ++i) o[dt][i] *= al[i];

    // ---- P: C-layout -> A-layout via LDS (bf16) ----
#pragma unroll
    for (int nt = 0; nt < 4; ++nt)
#pragma unroll
      for (int i = 0; i < 8; ++i)
        P[(hs * 8 + i) * 80 + nt * 16 + (lane & 15)] = (__bf16)s4[nt][i];
    asm volatile("s_wait_dscnt 0" ::: "memory");

    v16bf pf0 = a_frag_bf16(P, 80, 0, 0);
    v16bf pf1 = a_frag_bf16(P, 80, 0, 32);

    // ---- O += P V  (V stored transposed [hd, T] -> contiguous B frags) ----
#pragma unroll
    for (int dt = 0; dt < 4; ++dt) {
      v16bf bv0 = b_frag_bf16T(Vh, SEQ, dt * 16, j0);
      v16bf bv1 = b_frag_bf16T(Vh, SEQ, dt * 16, j0 + 32);
      o[dt] = wmma_bf16(pf0, bv0, o[dt]);
      o[dt] = wmma_bf16(pf1, bv1, o[dt]);
    }
  }

  // ---- normalize and store attn output bf16 as [B*T, DIM], col = h*64+d ----
  float inv[8];
#pragma unroll
  for (int i = 0; i < 8; ++i) inv[i] = 1.0f / l[i];
#pragma unroll
  for (int dt = 0; dt < 4; ++dt)
#pragma unroll
    for (int i = 0; i < 8; ++i) {
      const int t = q0 + hs * 8 + i;
      AO[((size_t)b * SEQ + t) * DIM + h * HD + dt * 16 + (lane & 15)] =
          (__bf16)(o[dt][i] * inv[i]);
    }
}

// ---------------------------------------------------------------------------
// Kernel 3: out = attn_out(bf16) * Woutbf^T  -> fp32 d_out [4096, 1024]
// Wave tile 32x128; WG = 4 waves covering 64x256.
// ---------------------------------------------------------------------------
__global__ __launch_bounds__(128) void out_gemm(
    const __bf16* __restrict__ AO, const __bf16* __restrict__ Wout,
    float* __restrict__ out) {
  const int w    = threadIdx.x >> 5;
  const int lane = threadIdx.x & 31;
  const int r0 = blockIdx.x * 64 + (w & 1) * 32;
  const int c0 = blockIdx.y * 256 + (w >> 1) * 128;

  const v8f zero = {0.f, 0.f, 0.f, 0.f, 0.f, 0.f, 0.f, 0.f};
  v8f acc[2][8];
#pragma unroll
  for (int rt = 0; rt < 2; ++rt)
#pragma unroll
    for (int ct = 0; ct < 8; ++ct) acc[rt][ct] = zero;

  for (int k0 = 0; k0 < DIM; k0 += 32) {
    v16bf a0 = a_frag_bf16(AO, DIM, r0, k0);
    v16bf a1 = a_frag_bf16(AO, DIM, r0 + 16, k0);
#pragma unroll
    for (int ct = 0; ct < 8; ++ct) {
      v16bf bb = b_frag_bf16T(Wout, DIM, c0 + ct * 16, k0);
      acc[0][ct] = wmma_bf16(a0, bb, acc[0][ct]);
      acc[1][ct] = wmma_bf16(a1, bb, acc[1][ct]);
    }
  }

  const int hs = lane >> 4;
#pragma unroll
  for (int rt = 0; rt < 2; ++rt)
#pragma unroll
    for (int ct = 0; ct < 8; ++ct) {
      const int n = c0 + ct * 16 + (lane & 15);
#pragma unroll
      for (int i = 0; i < 8; ++i) {
        const int t = r0 + rt * 16 + hs * 8 + i;
        out[(size_t)t * DIM + n] = acc[rt][ct][i];
      }
    }
}

// ---------------------------------------------------------------------------
extern "C" void kernel_launch(void* const* d_in, const int* in_sizes, int n_in,
                              void* d_out, int out_size, void* d_ws, size_t ws_size,
                              hipStream_t stream) {
  const float* x    = (const float*)d_in[0];   // [2, 2048, 1024]
  const float* Wqkv = (const float*)d_in[1];   // [3072, 1024]
  const float* Wout = (const float*)d_in[2];   // [1024, 1024]
  float* out = (float*)d_out;                  // [2, 2048, 1024] fp32

  char* ws = (char*)d_ws;
  const size_t MB = (size_t)1024 * 1024;
  __bf16* Qb    = (__bf16*)(ws + 0 * MB);      // [B,H,T,hd]   8 MB
  __bf16* Kb    = (__bf16*)(ws + 8 * MB);      // [B,H,T,hd]   8 MB
  __bf16* Vt    = (__bf16*)(ws + 16 * MB);     // [B,H,hd,T]   8 MB
  __bf16* AO    = (__bf16*)(ws + 24 * MB);     // [B,T,dim]    8 MB
  __bf16* Xbf   = (__bf16*)(ws + 32 * MB);     // [B*T,dim]    8 MB
  __bf16* Wqkvb = (__bf16*)(ws + 40 * MB);     // [3072,1024]  6 MB
  __bf16* Woutb = (__bf16*)(ws + 46 * MB);     // [1024,1024]  2 MB

  // one-shot precision conversion (all sizes divisible by 2048 elems/block)
  cvt_f32_to_bf16<<<2048, 256, 0, stream>>>(x,    Xbf,   BATCH * SEQ * DIM);
  cvt_f32_to_bf16<<<1536, 256, 0, stream>>>(Wqkv, Wqkvb, 3 * DIM * DIM);
  cvt_f32_to_bf16<<<512,  256, 0, stream>>>(Wout, Woutb, DIM * DIM);

  qkv_gemm<<<dim3(64, 12), 128, 0, stream>>>(Xbf, Wqkvb, Qb, Kb, Vt);
  attn_kernel<<<dim3(32, 16, 2), 128, 0, stream>>>(Qb, Kb, Vt, AO);
  out_gemm<<<dim3(64, 4), 128, 0, stream>>>(AO, Woutb, out);
}
